// LayerAdapter_38869454029673
// MI455X (gfx1250) — matrix operand order repaired and checked by reading, not compiled, not run on hardware
//
#include <hip/hip_runtime.h>
#include <hip/hip_bf16.h>
#include <math.h>
#include <stdint.h>

typedef __bf16 bf16;
typedef __attribute__((ext_vector_type(16))) __bf16 v16bf;
typedef __attribute__((ext_vector_type(8)))  float  v8f;

#define D_MODEL 768
#define L_LAYERS 12
#define R_RANK 8
#define BT 4096            // B*T
#define MKV (BT * L_LAYERS)

// ---------------------------------------------------------------------------
// fp32 -> bf16 weight conversion (one-shot, tiny)
// ---------------------------------------------------------------------------
__global__ void cvt_f32_bf16(const float* __restrict__ src, bf16* __restrict__ dst, int n) {
  int i = blockIdx.x * blockDim.x + threadIdx.x;
  if (i < n) dst[i] = (bf16)src[i];
}

// ---------------------------------------------------------------------------
// Q = hidden[..., L-1]  ->  bf16 compact [BT][768]
// ---------------------------------------------------------------------------
__global__ void prep_q(const float* __restrict__ hidden, bf16* __restrict__ qin) {
  int i = blockIdx.x * blockDim.x + threadIdx.x;   // over BT*768
  qin[i] = (bf16)hidden[(size_t)i * L_LAYERS + (L_LAYERS - 1)];
}

// ---------------------------------------------------------------------------
// kv[bt,l,:] = hidden[bt,:,l] + ( (x_l @ B_l^T) @ A_l^T + b_l ),  x_l = premlp[bt,:,l]
// one wave32 per (bt,l); rank-8 partials reduced with __shfl_xor
// ---------------------------------------------------------------------------
__global__ __launch_bounds__(256) void prep_kv(const float* __restrict__ hidden,
                                               const float* __restrict__ premlp,
                                               const float* __restrict__ A_lr,
                                               const float* __restrict__ B_lr,
                                               const float* __restrict__ b_lr,
                                               bf16* __restrict__ kv) {
  const int lane = threadIdx.x & 31;
  const int gw   = blockIdx.x * (blockDim.x >> 5) + (threadIdx.x >> 5);  // bt*L + l
  const int l    = gw % L_LAYERS;
  const int bt   = gw / L_LAYERS;

  const float* x  = premlp + (size_t)bt * D_MODEL * L_LAYERS + l;
  const float* Bl = B_lr + (size_t)l * R_RANK * D_MODEL;

  float t[R_RANK];
#pragma unroll
  for (int r = 0; r < R_RANK; ++r) t[r] = 0.f;

  for (int d = lane; d < D_MODEL; d += 32) {
    float xv = x[(size_t)d * L_LAYERS];
#pragma unroll
    for (int r = 0; r < R_RANK; ++r) t[r] += xv * Bl[r * D_MODEL + d];
  }
#pragma unroll
  for (int r = 0; r < R_RANK; ++r) {
#pragma unroll
    for (int off = 16; off > 0; off >>= 1) t[r] += __shfl_xor(t[r], off, 32);
  }

  const float* Al = A_lr + (size_t)l * D_MODEL * R_RANK;
  const float* bl = b_lr + (size_t)l * D_MODEL;
  const float* hd = hidden + (size_t)bt * D_MODEL * L_LAYERS + l;
  bf16* out = kv + (size_t)gw * D_MODEL;

  for (int d = lane; d < D_MODEL; d += 32) {
    float p = bl[d];
#pragma unroll
    for (int r = 0; r < R_RANK; ++r) p += t[r] * Al[(size_t)d * R_RANK + r];
    out[d] = (bf16)(hd[(size_t)d * L_LAYERS] + p);
  }
}

// ---------------------------------------------------------------------------
// per-thread async stage of one 32B slice of each 128x32 bf16 tile
// (INST_OFFSET applies to both the LDS and the global address)
// ---------------------------------------------------------------------------
__device__ __forceinline__ void async_tile(unsigned ldsA, unsigned ldsB,
                                           const bf16* ga, const bf16* gb) {
  asm volatile(
      "global_load_async_to_lds_b128 %0, %2, off\n\t"
      "global_load_async_to_lds_b128 %0, %2, off offset:16\n\t"
      "global_load_async_to_lds_b128 %1, %3, off\n\t"
      "global_load_async_to_lds_b128 %1, %3, off offset:16"
      :
      : "v"(ldsA), "v"(ldsB),
        "v"((unsigned long long)(uintptr_t)ga),
        "v"((unsigned long long)(uintptr_t)gb)
      : "memory");
}

// ---------------------------------------------------------------------------
// C[M,768] = A[M,768] (bf16) @ W[768,768]^T (bf16, row-major N,K) + bias
// block tile: 128(M) x 128(N), K-step 32, 8 waves, double-buffered LDS.
// Each wave: one 16x32 A fragment reused across 8 B tiles -> 8 WMMAs/K-step.
// Staging: global_load_async_to_lds_b128; next tile kept in flight behind the
// WMMAs via partial s_wait_asynccnt 4 (async loads complete in order).
// mode 0: outb[row*768+col] = bf16(acc + bias)
// mode 1: outf[row*768+col] = resid[(row*768+col)*12 + 11] + scale*(acc + bias)
// ---------------------------------------------------------------------------
__global__ __launch_bounds__(256) void gemm_bf16(const bf16* __restrict__ A,
                                                 const bf16* __restrict__ W,
                                                 const float* __restrict__ bias,
                                                 bf16* __restrict__ outb,
                                                 float* __restrict__ outf,
                                                 const float* __restrict__ resid,
                                                 const float* __restrict__ scale_ptr,
                                                 int mode) {
  constexpr int LDT  = 40;          // padded LDS row stride (bf16) = 80B
  constexpr int TILE = 128 * LDT;   // one 128x32 tile (padded), in bf16 units
  __shared__ __align__(16) bf16 As[2 * TILE];
  __shared__ __align__(16) bf16 Bs[2 * TILE];

  const int tid  = threadIdx.x;
  const int lane = tid & 31;
  const int wave = tid >> 5;
  const int half = lane >> 4;
  const int l16  = lane & 15;
  const int M0   = blockIdx.x * 128;
  const int N0   = blockIdx.y * 128;

  v8f acc[8] = {};

  // cooperative-load geometry: 256 threads, each owns 32B of a 128x32 bf16 tile
  const int row  = tid >> 1;   // 0..127
  const int part = tid & 1;    // 0..1 (16 bf16 each)
  const bf16* gA = A + (size_t)(M0 + row) * D_MODEL + part * 16;
  const bf16* gB = W + (size_t)(N0 + row) * D_MODEL + part * 16;
  const unsigned ldsA[2] = {
      (unsigned)(uintptr_t)(As + row * LDT + part * 16),
      (unsigned)(uintptr_t)(As + TILE + row * LDT + part * 16)};
  const unsigned ldsB[2] = {
      (unsigned)(uintptr_t)(Bs + row * LDT + part * 16),
      (unsigned)(uintptr_t)(Bs + TILE + row * LDT + part * 16)};

  // prologue: stage tile 0 into buffer 0
  async_tile(ldsA[0], ldsB[0], gA, gB);

  for (int k0 = 0; k0 < D_MODEL; k0 += 32) {
    const int cur = (k0 >> 5) & 1;
    if (k0 + 32 < D_MODEL) {
      // stage tile k+1 into the other buffer; keep it in flight behind compute
      async_tile(ldsA[cur ^ 1], ldsB[cur ^ 1], gA + k0 + 32, gB + k0 + 32);
      asm volatile("s_wait_asynccnt 4" ::: "memory");  // tile k done, k+1 in flight
    } else {
      asm volatile("s_wait_asynccnt 0" ::: "memory");
    }
    __syncthreads();   // publish tile k to all waves

    const bf16* Ac = As + cur * TILE;
    const bf16* Bc = Bs + cur * TILE;

    // A fragment: row = wave*16 + l16; lanes 0-15: K {0..7,16..23}, 16-31: +8
    union { uint4 u[2]; v16bf v; } fa;
    {
      const uint4* ap = (const uint4*)(Ac + (wave * 16 + l16) * LDT);
      fa.u[0] = ap[half];       // K = half*8 .. half*8+7
      fa.u[1] = ap[2 + half];   // K = 16+half*8 .. +7
    }
#pragma unroll
    for (int nt = 0; nt < 8; ++nt) {
      // B fragment: N = nt*16 + l16; lanes 0-15: K {0..7,16..23}, 16-31: +8
      union { uint4 u[2]; v16bf v; } fb;
      const uint4* bp = (const uint4*)(Bc + (nt * 16 + l16) * LDT);
      fb.u[0] = bp[half];
      fb.u[1] = bp[2 + half];
      acc[nt] = __builtin_amdgcn_wmma_f32_16x16x32_bf16(
          false, fa.v, false, fb.v, (short)0, acc[nt], false, false);
    }
    __syncthreads();   // all waves done reading tile k before it is overwritten
  }

  const float scale = (mode == 1) ? scale_ptr[0] : 0.f;
#pragma unroll
  for (int nt = 0; nt < 8; ++nt) {
    const int col = N0 + nt * 16 + l16;
    const float bb = bias[col];
#pragma unroll
    for (int r = 0; r < 8; ++r) {
      const int rw = M0 + wave * 16 + r + 8 * half;   // C layout: VGPR r -> M=r / M=8+r
      const float val = acc[nt][r] + bb;
      if (mode == 0) {
        outb[(size_t)rw * D_MODEL + col] = (bf16)val;
      } else {
        const float q = resid[((size_t)rw * D_MODEL + col) * L_LAYERS + (L_LAYERS - 1)];
        outf[(size_t)rw * D_MODEL + col] = q + scale * val;
      }
    }
  }
}

// ---------------------------------------------------------------------------
// attention over L=12 layers per (bt, head); one wave32 per (bt,h),
// each lane owns 2 of the 64 head dims
// ---------------------------------------------------------------------------
__global__ __launch_bounds__(256) void attn_kernel(const bf16* __restrict__ q,
                                                   const bf16* __restrict__ kp,
                                                   const bf16* __restrict__ vp,
                                                   bf16* __restrict__ attn_out) {
  const int lane = threadIdx.x & 31;
  const int gw   = blockIdx.x * (blockDim.x >> 5) + (threadIdx.x >> 5);  // bt*12 + h
  const int h    = gw % 12;
  const int bt   = gw / 12;

  const bf16* qh = q + (size_t)bt * D_MODEL + h * 64;
  const float q0 = (float)qh[lane * 2];
  const float q1 = (float)qh[lane * 2 + 1];

  const bf16* kb = kp + (size_t)bt * L_LAYERS * D_MODEL + h * 64;
  float sc[L_LAYERS];
#pragma unroll
  for (int l = 0; l < L_LAYERS; ++l) {
    const bf16* kl = kb + (size_t)l * D_MODEL;
    float s = q0 * (float)kl[lane * 2] + q1 * (float)kl[lane * 2 + 1];
#pragma unroll
    for (int off = 16; off > 0; off >>= 1) s += __shfl_xor(s, off, 32);
    sc[l] = s * 0.125f;  // 1/sqrt(64)
  }
  float mx = sc[0];
#pragma unroll
  for (int l = 1; l < L_LAYERS; ++l) mx = fmaxf(mx, sc[l]);
  float sum = 0.f;
#pragma unroll
  for (int l = 0; l < L_LAYERS; ++l) { sc[l] = __expf(sc[l] - mx); sum += sc[l]; }
  const float inv = 1.f / sum;

  const bf16* vb = vp + (size_t)bt * L_LAYERS * D_MODEL + h * 64;
  float a0 = 0.f, a1 = 0.f;
#pragma unroll
  for (int l = 0; l < L_LAYERS; ++l) {
    const bf16* vl = vb + (size_t)l * D_MODEL;
    a0 += sc[l] * (float)vl[lane * 2];
    a1 += sc[l] * (float)vl[lane * 2 + 1];
  }
  bf16* o = attn_out + (size_t)bt * D_MODEL + h * 64 + lane * 2;
  o[0] = (bf16)(a0 * inv);
  o[1] = (bf16)(a1 * inv);
}

// ---------------------------------------------------------------------------
extern "C" void kernel_launch(void* const* d_in, const int* in_sizes, int n_in,
                              void* d_out, int out_size, void* d_ws, size_t ws_size,
                              hipStream_t stream) {
  const float* hidden = (const float*)d_in[0];
  const float* premlp = (const float*)d_in[1];
  const float* A_lr   = (const float*)d_in[2];
  const float* B_lr   = (const float*)d_in[3];
  const float* b_lr   = (const float*)d_in[4];
  const float* Wq     = (const float*)d_in[5];
  const float* bq     = (const float*)d_in[6];
  const float* Wk     = (const float*)d_in[7];
  const float* bk     = (const float*)d_in[8];
  const float* Wv     = (const float*)d_in[9];
  const float* bv     = (const float*)d_in[10];
  const float* Wo     = (const float*)d_in[11];
  const float* bo     = (const float*)d_in[12];
  const float* scale  = (const float*)d_in[13];

  char* ws = (char*)d_ws;
  size_t off = 0;
  auto alloc = [&](size_t bytes) -> void* {
    void* p = ws + off;
    off += (bytes + 255) & ~(size_t)255;
    return p;
  };
  const size_t WSZ = (size_t)D_MODEL * D_MODEL * sizeof(bf16);
  bf16* wq_b   = (bf16*)alloc(WSZ);
  bf16* wk_b   = (bf16*)alloc(WSZ);
  bf16* wv_b   = (bf16*)alloc(WSZ);
  bf16* wo_b   = (bf16*)alloc(WSZ);
  bf16* kv_b   = (bf16*)alloc((size_t)MKV * D_MODEL * sizeof(bf16));
  bf16* kp_b   = (bf16*)alloc((size_t)MKV * D_MODEL * sizeof(bf16));
  bf16* vp_b   = (bf16*)alloc((size_t)MKV * D_MODEL * sizeof(bf16));
  bf16* qin_b  = (bf16*)alloc((size_t)BT * D_MODEL * sizeof(bf16));
  bf16* qpr_b  = (bf16*)alloc((size_t)BT * D_MODEL * sizeof(bf16));
  bf16* attn_b = (bf16*)alloc((size_t)BT * D_MODEL * sizeof(bf16));

  const int nW = D_MODEL * D_MODEL;  // 589824
  cvt_f32_bf16<<<nW / 256, 256, 0, stream>>>(Wq, wq_b, nW);
  cvt_f32_bf16<<<nW / 256, 256, 0, stream>>>(Wk, wk_b, nW);
  cvt_f32_bf16<<<nW / 256, 256, 0, stream>>>(Wv, wv_b, nW);
  cvt_f32_bf16<<<nW / 256, 256, 0, stream>>>(Wo, wo_b, nW);

  prep_q<<<(BT * D_MODEL) / 256, 256, 0, stream>>>(hidden, qin_b);
  prep_kv<<<MKV / 8, 256, 0, stream>>>(hidden, premlp, A_lr, B_lr, b_lr, kv_b);

  // Q projection: M = 4096
  gemm_bf16<<<dim3(BT / 128, D_MODEL / 128), 256, 0, stream>>>(
      qin_b, wq_b, bq, qpr_b, nullptr, nullptr, nullptr, 0);
  // K / V projections: M = 49152 (the 116-GFLOP workhorses)
  gemm_bf16<<<dim3(MKV / 128, D_MODEL / 128), 256, 0, stream>>>(
      kv_b, wk_b, bk, kp_b, nullptr, nullptr, nullptr, 0);
  gemm_bf16<<<dim3(MKV / 128, D_MODEL / 128), 256, 0, stream>>>(
      kv_b, wv_b, bv, vp_b, nullptr, nullptr, nullptr, 0);

  attn_kernel<<<(BT * 12) / 8, 256, 0, stream>>>(qpr_b, kp_b, vp_b, attn_b);

  // O projection + fused residual epilogue -> fp32 d_out
  gemm_bf16<<<dim3(BT / 128, D_MODEL / 128), 256, 0, stream>>>(
      attn_b, wo_b, bo, nullptr, (float*)d_out, hidden, scale, 1);
}
